// CharRNN_86792699118058
// MI455X (gfx1250) — matrix operand order repaired and checked by reading, not compiled, not run on hardware
//
#include <hip/hip_runtime.h>

// Problem constants (match reference)
#define BB 256
#define LL 512
#define VV 256
#define EE 256
#define HH 1024

typedef __attribute__((ext_vector_type(16))) __bf16 v16bf;
typedef __attribute__((ext_vector_type(8)))  __bf16 v8bf;
typedef __attribute__((ext_vector_type(8)))  float  v8f;

// ---------------------------------------------------------------------------
// Fragment loaders, contiguous k order shared by A and B (not the ISA table order)
// A (16x32 bf16, row-major, K contiguous): lane = 16*h + r, r = row (M),
//   VGPR0-7 hold K = 16h+0..15                             -> two 16B loads.
// B (32x16 bf16) sourced from a pre-transposed BT (N-major, K contiguous):
//   lane = 16*h + n, VGPR0-7 hold K = 16h+0..15            -> one 32B load.
// ---------------------------------------------------------------------------
__device__ __forceinline__ v16bf cat8(v8bf lo, v8bf hi) {
  return __builtin_shufflevector(lo, hi, 0,1,2,3,4,5,6,7,8,9,10,11,12,13,14,15);
}

__device__ __forceinline__ v16bf load_a_frag(const __bf16* row, int k0) {
  const int h = (threadIdx.x >> 4) & 1;
  v8bf lo = *(const v8bf*)(row + k0 + 16 * h);
  v8bf hi = *(const v8bf*)(row + k0 + 16 * h + 8);
  return cat8(lo, hi);
}

__device__ __forceinline__ v16bf load_b_frag(const __bf16* colrow, int k0) {
  const int h = (threadIdx.x >> 4) & 1;
  const v8bf* p = (const v8bf*)(colrow + k0 + 16 * h);
  return cat8(p[0], p[1]);
}

__device__ __forceinline__ v8f wmma_bf16(v16bf a, v16bf b, v8f c) {
  return __builtin_amdgcn_wmma_f32_16x16x32_bf16(false, a, false, b, (short)0, c,
                                                 false, false);
}

// One 16x16 f32 tile, K-reduction with bf16 WMMA, f32 accumulation.
__device__ __forceinline__ v8f wmma_tile(const __bf16* arow, const __bf16* bcol, int K) {
  v8f c = {};
#pragma unroll 8
  for (int k = 0; k < K; k += 32) {
    c = wmma_bf16(load_a_frag(arow, k), load_b_frag(bcol, k), c);
  }
  return c;
}

// ---------------------------------------------------------------------------
// Prep: convert/transpose weights to bf16 once; zero barrier counters.
// ---------------------------------------------------------------------------
__global__ void prep_kernel(const float* __restrict__ hidden,
                            const float* __restrict__ emb,
                            const float* __restrict__ We,
                            const float* __restrict__ Wh,
                            const float* __restrict__ Wo,
                            __bf16* __restrict__ emb_bf,
                            __bf16* __restrict__ WeT,
                            __bf16* __restrict__ WhT,
                            __bf16* __restrict__ WoT,
                            __bf16* __restrict__ h_init,
                            int* __restrict__ cnt) {
  int j = blockIdx.x * blockDim.x + threadIdx.x;
  const int nEmb = VV * EE;   // 65536
  const int nWeT = HH * EE;   // 262144
  const int nWhT = HH * HH;   // 1048576
  const int nWoT = VV * HH;   // 262144
  const int nHin = BB * HH;   // 262144
  if (j < nEmb) { emb_bf[j] = (__bf16)emb[j]; return; }
  j -= nEmb;
  if (j < nWeT) { int n = j / EE, e = j % EE; WeT[j] = (__bf16)We[e * HH + n]; return; }
  j -= nWeT;
  if (j < nWhT) { int n = j / HH, k = j % HH; WhT[j] = (__bf16)Wh[k * HH + n]; return; }
  j -= nWhT;
  if (j < nWoT) { int v = j / HH, k = j % HH; WoT[j] = (__bf16)Wo[k * VV + v]; return; }
  j -= nWoT;
  if (j < nHin) { h_init[j] = (__bf16)hidden[j]; return; }
  j -= nHin;
  if (j < 16) { cnt[j] = 0; }
}

// ---------------------------------------------------------------------------
// Input projection: xproj[l,b,:] = emb[x[b,l]] @ W_e   (f32 result)
// tiles = (L * B/16) * (H/16) = 8192 * 64 = 524288 -> 65536 blocks x 8 waves.
// ---------------------------------------------------------------------------
__global__ void embed_gemm_kernel(const int* __restrict__ x,
                                  const __bf16* __restrict__ emb_bf,
                                  const __bf16* __restrict__ WeT,
                                  float* __restrict__ xproj) {
  const int wave = threadIdx.x >> 5;
  const int lane = threadIdx.x & 31;
  const int tile = blockIdx.x * 8 + wave;
  const int mt = tile >> 6;          // 0..8191 : (l, b-tile)
  const int nt = tile & 63;          // 0..63   : H/16
  const int l  = mt >> 4;
  const int b0 = (mt & 15) * 16;

  const int r  = lane & 15;
  const int hl = lane >> 4;

  const int idx = x[(b0 + r) * LL + l];              // gather embedding row
  const __bf16* arow = emb_bf + (size_t)idx * EE;
  const __bf16* bcol = WeT + (size_t)(nt * 16 + r) * EE;

  v8f c = wmma_tile(arow, bcol, EE);

  float* outp = xproj + (size_t)l * BB * HH;
#pragma unroll
  for (int i = 0; i < 8; ++i) {
    int m = b0 + hl * 8 + i;
    int n = nt * 16 + r;             // lane&15 == local N for C/D layout
    outp[(size_t)m * HH + n] = c[i];
  }
}

// ---------------------------------------------------------------------------
// Persistent recurrence: one launch runs all 512 steps.
//   h_t = tanh(xproj[t] + h_{t-1} @ W_h)
// 1024 tiles (B/16 x H/16) -> 128 blocks x 8 waves; each wave owns one
// (mt,nt) tile for the whole sequence and keeps its 32 W_hT K-fragments
// (256 VGPRs) resident in registers across all steps.
// __launch_bounds__(256, 1): full VGPR budget, no spill of bfrag.
// Inter-step dependency: group mt (8 consecutive blocks) only consumes rows
// produced by group mt -> monotonic atomic-counter barrier over 8 blocks.
// ---------------------------------------------------------------------------
__global__ void __launch_bounds__(256, 1)
rnn_persistent_kernel(const __bf16* __restrict__ h_init,
                      const __bf16* __restrict__ WhT,
                      const float* __restrict__ xproj,
                      __bf16* __restrict__ outs,
                      float* __restrict__ final_hidden,
                      int* __restrict__ cnt) {
  const int wave = threadIdx.x >> 5;
  const int lane = threadIdx.x & 31;
  const int tile = blockIdx.x * 8 + wave;   // 0..1023
  const int mt = tile >> 6;                 // 0..15  (B/16) ; == blockIdx.x>>3
  const int nt = tile & 63;                 // 0..63  (H/16)
  const int r  = lane & 15;
  const int hl = lane >> 4;

  // Preload this tile's W_hT fragments: loop-invariant across all 512 steps.
  const __bf16* bcol = WhT + (size_t)(nt * 16 + r) * HH;
  v16bf bfrag[32];
#pragma unroll
  for (int k = 0; k < 32; ++k) bfrag[k] = load_b_frag(bcol, k * 32);

  int* my_cnt = cnt + (blockIdx.x >> 3);
  const __bf16* hprev = h_init;

  for (int t = 0; t < LL; ++t) {
    const float* xproj_t = xproj + (size_t)t * BB * HH;
    __bf16* hout = outs + (size_t)t * BB * HH;
    const __bf16* arow = hprev + (size_t)(mt * 16 + r) * HH;

    // Prefetch the f32 bias tile we will add in the epilogue.
    __builtin_prefetch(xproj_t + (size_t)(mt * 16 + hl * 8) * HH + nt * 16 + r, 0, 1);

    v8f c = {};
#pragma unroll
    for (int k = 0; k < 32; ++k) {
      c = wmma_bf16(load_a_frag(arow, k * 32), bfrag[k], c);
    }

#pragma unroll
    for (int i = 0; i < 8; ++i) {
      int m = mt * 16 + hl * 8 + i;
      int n = nt * 16 + r;
      float pre = c[i] + xproj_t[(size_t)m * HH + n];
      float y = tanhf(pre);
      hout[(size_t)m * HH + n] = (__bf16)y;
      if (t == LL - 1) final_hidden[(size_t)m * HH + n] = y;
    }

    // ---- 8-block group barrier (release/acquire at agent scope) ----
    __builtin_amdgcn_fence(__ATOMIC_RELEASE, "agent");  // drain h_t stores
    __syncthreads();
    if (threadIdx.x == 0) {
      __hip_atomic_fetch_add(my_cnt, 1, __ATOMIC_RELAXED, __HIP_MEMORY_SCOPE_AGENT);
      while (__hip_atomic_load(my_cnt, __ATOMIC_RELAXED, __HIP_MEMORY_SCOPE_AGENT) <
             8 * (t + 1)) {
        __builtin_amdgcn_s_sleep(1);
      }
    }
    __syncthreads();
    __builtin_amdgcn_fence(__ATOMIC_ACQUIRE, "agent");  // see other WGPs' h_t

    hprev = hout;
  }
}

// ---------------------------------------------------------------------------
// Logits: logits[b,l,:] = outs[l,b,:] @ W_o, (l,b)->(b,l) permute fused in
// the store. tiles = 8192 * 16 = 131072 -> 16384 blocks x 8 waves.
// ---------------------------------------------------------------------------
__global__ void out_gemm_kernel(const __bf16* __restrict__ outs,
                                const __bf16* __restrict__ WoT,
                                float* __restrict__ logits) {
  const int wave = threadIdx.x >> 5;
  const int lane = threadIdx.x & 31;
  const int tile = blockIdx.x * 8 + wave;
  const int mt = tile >> 4;          // 0..8191 : (l, b-tile)
  const int nt = tile & 15;          // 0..15   : V/16
  const int l  = mt >> 4;
  const int b0 = (mt & 15) * 16;

  const int r  = lane & 15;
  const int hl = lane >> 4;

  const __bf16* arow = outs + (size_t)l * BB * HH + (size_t)(b0 + r) * HH;
  const __bf16* bcol = WoT + (size_t)(nt * 16 + r) * HH;

  v8f c = wmma_tile(arow, bcol, HH);

#pragma unroll
  for (int i = 0; i < 8; ++i) {
    int b = b0 + hl * 8 + i;
    int v = nt * 16 + r;
    logits[((size_t)b * LL + l) * VV + v] = c[i];
  }
}

// ---------------------------------------------------------------------------
extern "C" void kernel_launch(void* const* d_in, const int* in_sizes, int n_in,
                              void* d_out, int out_size, void* d_ws, size_t ws_size,
                              hipStream_t stream) {
  const int*   x      = (const int*)d_in[0];     // [B, L] (jax default: int32)
  const float* hidden = (const float*)d_in[1];   // [B, H]
  const float* emb    = (const float*)d_in[2];   // [V, E]
  const float* We     = (const float*)d_in[3];   // [E, H]
  const float* Wh     = (const float*)d_in[4];   // [H, H]
  const float* Wo     = (const float*)d_in[5];   // [H, V]
  float* out = (float*)d_out;                    // logits [B,L,V] ++ final_hidden [B,H]

  // Workspace layout (all offsets naturally aligned):
  char* ws = (char*)d_ws;
  size_t off = 0;
  float*  xproj  = (float*)(ws + off);  off += (size_t)LL * BB * HH * 4;  // 512 MB
  __bf16* outs   = (__bf16*)(ws + off); off += (size_t)LL * BB * HH * 2;  // 256 MB
  __bf16* h_init = (__bf16*)(ws + off); off += (size_t)BB * HH * 2;
  __bf16* emb_bf = (__bf16*)(ws + off); off += (size_t)VV * EE * 2;
  __bf16* WeT    = (__bf16*)(ws + off); off += (size_t)HH * EE * 2;
  __bf16* WhT    = (__bf16*)(ws + off); off += (size_t)HH * HH * 2;
  __bf16* WoT    = (__bf16*)(ws + off); off += (size_t)VV * HH * 2;
  int*    cnt    = (int*)(ws + off);    off += 256;
  (void)in_sizes; (void)n_in; (void)out_size; (void)ws_size;

  // 1) weight conversion / transposition + barrier-counter reset
  const int prepN = VV * EE + HH * EE + HH * HH + VV * HH + BB * HH + 16;
  prep_kernel<<<(prepN + 255) / 256, 256, 0, stream>>>(hidden, emb, We, Wh, Wo,
                                                       emb_bf, WeT, WhT, WoT,
                                                       h_init, cnt);

  // 2) batched input projection (gather + GEMM)
  embed_gemm_kernel<<<65536, 256, 0, stream>>>(x, emb_bf, WeT, xproj);

  // 3) full recurrence in ONE persistent launch (group-local atomic barrier)
  float* final_hidden = out + (size_t)BB * LL * VV;
  rnn_persistent_kernel<<<128, 256, 0, stream>>>(h_init, WhT, xproj, outs,
                                                 final_hidden, cnt);

  // 4) output projection with fused (l,b)->(b,l) permute
  out_gemm_kernel<<<16384, 256, 0, stream>>>(outs, WoT, out);
}
